// Qwen2MoeSparseMoeBlock_44659069944441
// MI455X (gfx1250) — matrix operand order repaired and checked by reading, not compiled, not run on hardware
//
#include <hip/hip_runtime.h>
#include <hip/hip_fp16.h>

typedef __attribute__((ext_vector_type(16))) _Float16 v16h;
typedef __attribute__((ext_vector_type(2)))  __fp16   v2fp16;  // cvt_pkrtz return type
typedef __attribute__((ext_vector_type(8)))  float    v8f;

#define TOK   8192   // B*S = 4*2048
#define HD    2048   // hidden
#define NE    16     // experts
#define TOPK  4
#define NI    1408   // expert intermediate
#define NIS   5632   // shared intermediate
#define CHUNK 2048   // shared-expert token chunk (CHUNK*NIS == TOK*NI)

// LDS fragment-order staging: each "slot" = 16 f16 (32B) padded to 48B (3 x uint4)
#define SLOT_U4 3

#define WMMA_F16(A, B, C) \
  __builtin_amdgcn_wmma_f32_16x16x32_f16(false, (A), false, (B), (short)0, (C), false, false)

// fast sigmoid via v_rcp_f32 (~1 ulp, far below f16 noise)
__device__ __forceinline__ float fast_sigmoid(float x) {
  return __builtin_amdgcn_rcpf(1.f + __expf(-x));
}

// ---------------- fp32 -> fp16 helpers (v_cvt_pk_rtz_f16_f32: 1 VALU per pair) ----
__device__ __forceinline__ unsigned pk2(float a, float b) {
  union { v2fp16 v; unsigned u; } c;
  c.v = __builtin_amdgcn_cvt_pkrtz(a, b);
  return c.u;
}

// fragment load: 32B (two ds_load_b128) from a 48B-strided slot
__device__ __forceinline__ v16h load_frag(const uint4* slot) {
  union { uint4 u[2]; v16h v; } c;
  c.u[0] = slot[0];
  c.u[1] = slot[1];
  return c.v;
}

// pack 16 fp32 (register-resident) -> one f16 slot (2 x 16B LDS stores)
__device__ __forceinline__ void pack_slot(const float* f, uint4* slot) {
  uint4 u0, u1;
  u0.x = pk2(f[0], f[1]);   u0.y = pk2(f[2], f[3]);
  u0.z = pk2(f[4], f[5]);   u0.w = pk2(f[6], f[7]);
  u1.x = pk2(f[8], f[9]);   u1.y = pk2(f[10], f[11]);
  u1.z = pk2(f[12], f[13]); u1.w = pk2(f[14], f[15]);
  slot[0] = u0; slot[1] = u1;
}

// ---------------- cast x to f16 ----------------
__global__ __launch_bounds__(256) void cast_x_kernel(const float* __restrict__ x,
                                                     _Float16* __restrict__ xb) {
  size_t i = ((size_t)blockIdx.x * 256 + threadIdx.x) * 4;
  float4 f = *(const float4*)(x + i);
  uint2 u; u.x = pk2(f.x, f.y); u.y = pk2(f.z, f.w);
  *(uint2*)((unsigned short*)xb + i) = u;
}

// ---------------- router: softmax + top4 scatter + shared gate ----------------
__global__ __launch_bounds__(256) void router_kernel(
    const float* __restrict__ x, const float* __restrict__ gw,
    const float* __restrict__ sgw, int* __restrict__ counts,
    int* __restrict__ list, float* __restrict__ wlist, float* __restrict__ gate) {
  int t = blockIdx.x;
  int tid = threadIdx.x;
  const float* xr = x + (size_t)t * HD;
  float acc[NE + 1];
#pragma unroll
  for (int e = 0; e <= NE; ++e) acc[e] = 0.f;
  for (int i = tid; i < HD; i += 256) {
    float xv = xr[i];
    const float* gr = gw + (size_t)i * NE;
#pragma unroll
    for (int e = 0; e < NE; ++e) acc[e] += xv * gr[e];
    acc[NE] += xv * sgw[i];
  }
  __shared__ float red[NE + 1][257];
#pragma unroll
  for (int e = 0; e <= NE; ++e) red[e][tid] = acc[e];
  __syncthreads();
  __shared__ float pv[NE + 1];
  if (tid <= NE) {
    float s = 0.f;
    for (int j = 0; j < 256; ++j) s += red[tid][j];
    pv[tid] = s;
  }
  __syncthreads();
  if (tid == 0) {
    float p[NE];
    float m = pv[0];
    for (int e = 1; e < NE; ++e) m = fmaxf(m, pv[e]);
    float s = 0.f;
    for (int e = 0; e < NE; ++e) { p[e] = __expf(pv[e] - m); s += p[e]; }
    float inv = __builtin_amdgcn_rcpf(s);
    for (int e = 0; e < NE; ++e) p[e] *= inv;
    for (int k = 0; k < TOPK; ++k) {
      int bj = 0; float bv = p[0];
      for (int e = 1; e < NE; ++e) if (p[e] > bv) { bv = p[e]; bj = e; }
      int pos = atomicAdd(&counts[bj], 1);
      list[bj * TOK + pos]  = t;
      wlist[bj * TOK + pos] = bv;
      p[bj] = -1.f;
    }
    gate[t] = fast_sigmoid(pv[NE]);
  }
}

// ---------------- fused gate/up GEMM: h = silu(x@W1) * (x@W3) -> f16 ----------------
// block tile 64(M) x 128(N), K-step 32, 8 waves each owning 32x32.
// Software-pipelined: tile kt+32 global loads issued while WMMAs consume tile kt.
template <bool GATHER>
__global__ __launch_bounds__(256) void gateup_kernel(
    const _Float16* __restrict__ xb, const float* __restrict__ w1,
    const float* __restrict__ w3, _Float16* __restrict__ hout,
    const int* __restrict__ rowlist, const int* __restrict__ countPtr,
    int countConst, int rowBase, int N) {
  int count = GATHER ? *countPtr : countConst;
  int m0 = blockIdx.x * 64;
  if (m0 >= count) return;
  int tid = threadIdx.x;
  int lane = tid & 31, wid = tid >> 5;
  int wm = (wid >> 2) * 32, wn = (wid & 3) * 32;
  int nb = blockIdx.y * 128;

  __shared__ int rows[64];
  __shared__ uint4 sA[128 * SLOT_U4];   // 4 row-tiles x 32 lane-slots
  __shared__ uint4 sB1[256 * SLOT_U4];  // 8 col-groups x 32 lane-slots
  __shared__ uint4 sB3[256 * SLOT_U4];

  if (tid < 64) {
    int m = m0 + tid; if (m > count - 1) m = count - 1;
    rows[tid] = GATHER ? rowlist[m] : (rowBase + m);
  }
  __syncthreads();

  // A writer: r = row in tile (0..63), s = 8-elem K segment (0..3)
  int ar = tid >> 2, as = tid & 3;
  const _Float16* arow = xb + (size_t)rows[ar] * HD + as * 8;
  uint4* aslot = &sA[(((ar >> 4) * 32) + (ar & 15) + ((as >> 1) << 4)) * SLOT_U4 + (as & 1)];
  // B writer: col (0..127), kh (0..1) -> slot (col/16)*32 + kh*16 + col%16
  int bcol = tid & 127, bkh = tid >> 7;
  const float* b1p = w1 + (size_t)(bkh * 16) * N + nb + bcol;
  const float* b3p = w3 + (size_t)(bkh * 16) * N + nb + bcol;
  uint4* b1slot = &sB1[(((bcol >> 4) * 32) + bkh * 16 + (bcol & 15)) * SLOT_U4];
  uint4* b3slot = &sB3[(((bcol >> 4) * 32) + bkh * 16 + (bcol & 15)) * SLOT_U4];
  // fragment slot pointers (loop-invariant)
  const uint4* a0p  = &sA[(((wm >> 4) + 0) * 32 + lane) * SLOT_U4];
  const uint4* a1p  = &sA[(((wm >> 4) + 1) * 32 + lane) * SLOT_U4];
  const uint4* bg0p = &sB1[(((wn >> 4) + 0) * 32 + lane) * SLOT_U4];
  const uint4* bg1p = &sB1[(((wn >> 4) + 1) * 32 + lane) * SLOT_U4];
  const uint4* bu0p = &sB3[(((wn >> 4) + 0) * 32 + lane) * SLOT_U4];
  const uint4* bu1p = &sB3[(((wn >> 4) + 1) * 32 + lane) * SLOT_U4];

  v8f accG[2][2], accU[2][2];
#pragma unroll
  for (int i = 0; i < 2; ++i)
#pragma unroll
    for (int j = 0; j < 2; ++j)
#pragma unroll
      for (int k = 0; k < 8; ++k) { accG[i][j][k] = 0.f; accU[i][j][k] = 0.f; }

  uint4 aReg;
  float fb1[16], fb3[16];
#define GU_LOAD_TILE(KT)                                           \
  {                                                                \
    aReg = *(const uint4*)(arow + (KT));                           \
    const float* p1 = b1p + (size_t)(KT) * N;                      \
    const float* p3 = b3p + (size_t)(KT) * N;                      \
    _Pragma("unroll") for (int j = 0; j < 16; ++j) {               \
      fb1[j] = p1[(size_t)j * N];                                  \
      fb3[j] = p3[(size_t)j * N];                                  \
    }                                                              \
  }

  GU_LOAD_TILE(0);
  for (int kt = 0; kt < HD; kt += 32) {
    *aslot = aReg;
    pack_slot(fb1, b1slot);
    pack_slot(fb3, b3slot);
    __syncthreads();
    if (kt + 32 < HD) GU_LOAD_TILE(kt + 32);   // overlap with WMMAs below
    if (kt + 64 < HD) {                        // warm L2 two tiles ahead
      __builtin_prefetch((const void*)(b1p + (size_t)(kt + 64) * N), 0, 0);
      __builtin_prefetch((const void*)(b3p + (size_t)(kt + 64) * N), 0, 0);
    }

    v16h a0  = load_frag(a0p);
    v16h a1  = load_frag(a1p);
    v16h bg0 = load_frag(bg0p);
    v16h bg1 = load_frag(bg1p);
    v16h bu0 = load_frag(bu0p);
    v16h bu1 = load_frag(bu1p);

    accG[0][0] = WMMA_F16(a0, bg0, accG[0][0]);
    accG[0][1] = WMMA_F16(a0, bg1, accG[0][1]);
    accG[1][0] = WMMA_F16(a1, bg0, accG[1][0]);
    accG[1][1] = WMMA_F16(a1, bg1, accG[1][1]);
    accU[0][0] = WMMA_F16(a0, bu0, accU[0][0]);
    accU[0][1] = WMMA_F16(a0, bu1, accU[0][1]);
    accU[1][0] = WMMA_F16(a1, bu0, accU[1][0]);
    accU[1][1] = WMMA_F16(a1, bu1, accU[1][1]);
    __syncthreads();
  }
#undef GU_LOAD_TILE

  int ro = (lane >> 4) << 3, co = lane & 15;
#pragma unroll
  for (int mi = 0; mi < 2; ++mi)
#pragma unroll
    for (int ni = 0; ni < 2; ++ni)
#pragma unroll
      for (int v = 0; v < 8; ++v) {
        int rit = wm + mi * 16 + v + ro;
        int m = m0 + rit;
        if (m < count) {
          int col = nb + wn + ni * 16 + co;
          float g = accG[mi][ni][v], u = accU[mi][ni][v];
          float h = g * fast_sigmoid(g) * u;   // silu(g)*u
          hout[(size_t)m * N + col] = (_Float16)h;
        }
      }
}

// ---------------- down projection: out[token] (+)= scale * (h @ W2) ----------------
template <bool GATHER, bool ACCUM>
__global__ __launch_bounds__(256) void down_kernel(
    const _Float16* __restrict__ hbuf, const float* __restrict__ w2,
    float* __restrict__ out, const int* __restrict__ rowlist,
    const float* __restrict__ wlist, const float* __restrict__ gate,
    const int* __restrict__ countPtr, int countConst, int rowBase, int Kd) {
  int count = GATHER ? *countPtr : countConst;
  int m0 = blockIdx.x * 64;
  if (m0 >= count) return;
  int tid = threadIdx.x;
  int lane = tid & 31, wid = tid >> 5;
  int wm = (wid >> 2) * 32, wn = (wid & 3) * 32;
  int nb = blockIdx.y * 128;

  __shared__ int toks[64];
  __shared__ float scs[64];
  __shared__ uint4 sA[128 * SLOT_U4];
  __shared__ uint4 sB[256 * SLOT_U4];

  if (tid < 64) {
    int m = m0 + tid; if (m > count - 1) m = count - 1;
    int tok = GATHER ? rowlist[m] : (rowBase + m);
    toks[tid] = tok;
    scs[tid]  = GATHER ? wlist[m] : gate[tok];
  }
  __syncthreads();

  int ar = tid >> 2, as = tid & 3;
  int am = m0 + ar; if (am > count - 1) am = count - 1;
  const _Float16* arow = hbuf + (size_t)am * Kd + as * 8;
  uint4* aslot = &sA[(((ar >> 4) * 32) + (ar & 15) + ((as >> 1) << 4)) * SLOT_U4 + (as & 1)];
  int bcol = tid & 127, bkh = tid >> 7;
  const float* bp = w2 + (size_t)(bkh * 16) * HD + nb + bcol;
  uint4* bslot = &sB[(((bcol >> 4) * 32) + bkh * 16 + (bcol & 15)) * SLOT_U4];
  const uint4* a0p = &sA[(((wm >> 4) + 0) * 32 + lane) * SLOT_U4];
  const uint4* a1p = &sA[(((wm >> 4) + 1) * 32 + lane) * SLOT_U4];
  const uint4* b0p = &sB[(((wn >> 4) + 0) * 32 + lane) * SLOT_U4];
  const uint4* b1p = &sB[(((wn >> 4) + 1) * 32 + lane) * SLOT_U4];

  v8f acc[2][2];
#pragma unroll
  for (int i = 0; i < 2; ++i)
#pragma unroll
    for (int j = 0; j < 2; ++j)
#pragma unroll
      for (int k = 0; k < 8; ++k) acc[i][j][k] = 0.f;

  uint4 aReg;
  float fb[16];
#define DN_LOAD_TILE(KT)                                           \
  {                                                                \
    aReg = *(const uint4*)(arow + (KT));                           \
    const float* p = bp + (size_t)(KT) * HD;                       \
    _Pragma("unroll") for (int j = 0; j < 16; ++j)                 \
      fb[j] = p[(size_t)j * HD];                                   \
  }

  DN_LOAD_TILE(0);
  for (int kt = 0; kt < Kd; kt += 32) {
    *aslot = aReg;
    pack_slot(fb, bslot);
    __syncthreads();
    if (kt + 32 < Kd) DN_LOAD_TILE(kt + 32);   // overlap with WMMAs below
    if (kt + 64 < Kd)                          // warm L2 two tiles ahead
      __builtin_prefetch((const void*)(bp + (size_t)(kt + 64) * HD), 0, 0);

    v16h a0 = load_frag(a0p);
    v16h a1 = load_frag(a1p);
    v16h b0 = load_frag(b0p);
    v16h b1 = load_frag(b1p);

    acc[0][0] = WMMA_F16(a0, b0, acc[0][0]);
    acc[0][1] = WMMA_F16(a0, b1, acc[0][1]);
    acc[1][0] = WMMA_F16(a1, b0, acc[1][0]);
    acc[1][1] = WMMA_F16(a1, b1, acc[1][1]);
    __syncthreads();
  }
#undef DN_LOAD_TILE

  int ro = (lane >> 4) << 3, co = lane & 15;
#pragma unroll
  for (int mi = 0; mi < 2; ++mi)
#pragma unroll
    for (int ni = 0; ni < 2; ++ni)
#pragma unroll
      for (int v = 0; v < 8; ++v) {
        int rit = wm + mi * 16 + v + ro;
        if (m0 + rit < count) {
          int col = nb + wn + ni * 16 + co;
          float val = scs[rit] * acc[mi][ni][v];
          size_t o = (size_t)toks[rit] * HD + col;
          if (ACCUM) out[o] += val; else out[o] = val;
        }
      }
}

// ---------------- host launcher ----------------
extern "C" void kernel_launch(void* const* d_in, const int* in_sizes, int n_in,
                              void* d_out, int out_size, void* d_ws, size_t ws_size,
                              hipStream_t stream) {
  const float* x   = (const float*)d_in[0];
  const float* gw  = (const float*)d_in[1];
  const float* w1  = (const float*)d_in[2];
  const float* w3  = (const float*)d_in[3];
  const float* w2  = (const float*)d_in[4];
  const float* sw1 = (const float*)d_in[5];
  const float* sw3 = (const float*)d_in[6];
  const float* sw2 = (const float*)d_in[7];
  const float* sgw = (const float*)d_in[8];
  float* out = (float*)d_out;

  char* ws = (char*)d_ws;
  size_t off = 0;
  auto alloc = [&](size_t bytes) {
    char* p = ws + off;
    off = (off + bytes + 255) & ~(size_t)255;
    return p;
  };
  _Float16* xb   = (_Float16*)alloc((size_t)TOK * HD * 2);     // 32 MB
  int*    counts = (int*)     alloc(NE * 4);
  int*    list   = (int*)     alloc((size_t)NE * TOK * 4);     // 512 KB
  float*  wlist  = (float*)   alloc((size_t)NE * TOK * 4);     // 512 KB
  float*  gate   = (float*)   alloc((size_t)TOK * 4);          // 32 KB
  _Float16* hs   = (_Float16*)alloc((size_t)TOK * NI * 2);     // 23 MB (== CHUNK*NIS*2)

  (void)hipMemsetAsync(counts, 0, NE * 4, stream);
  cast_x_kernel<<<(TOK * HD) / 1024, 256, 0, stream>>>(x, xb);
  router_kernel<<<TOK, 256, 0, stream>>>(x, gw, sgw, counts, list, wlist, gate);

  // Shared expert first: plain stores initialize out (no 64MB memset needed).
  for (int c = 0; c < TOK / CHUNK; ++c) {
    dim3 g1(CHUNK / 64, NIS / 128);
    gateup_kernel<false><<<g1, 256, 0, stream>>>(
        xb, sw1, sw3, hs, nullptr, nullptr, CHUNK, c * CHUNK, NIS);
    dim3 g2(CHUNK / 64, HD / 128);
    down_kernel<false, false><<<g2, 256, 0, stream>>>(
        hs, sw2, out, nullptr, nullptr, gate, nullptr, CHUNK, c * CHUNK, NIS);
  }
  // Routed experts: gather token lists, accumulate weighted down-proj into out.
  for (int e = 0; e < NE; ++e) {
    dim3 g1(TOK / 64, NI / 128);
    gateup_kernel<true><<<g1, 256, 0, stream>>>(
        xb, w1 + (size_t)e * HD * NI, w3 + (size_t)e * HD * NI, hs,
        list + (size_t)e * TOK, counts + e, 0, 0, NI);
    dim3 g2(TOK / 64, HD / 128);
    down_kernel<true, true><<<g2, 256, 0, stream>>>(
        hs, w2 + (size_t)e * NI * HD, out,
        list + (size_t)e * TOK, wlist + (size_t)e * TOK,
        nullptr, counts + e, 0, 0, NI);
  }
}
